// AthenaSA_83743272338098
// MI455X (gfx1250) — matrix operand order repaired and checked by reading, not compiled, not run on hardware
//
#include <hip/hip_runtime.h>
#include <hip/hip_bf16.h>

// ---------------------------------------------------------------------------
// Problem constants: B=2, S=2048, E=2048, H=16, K=128, V=128
// ---------------------------------------------------------------------------
#define SB   2048      // sequence
#define EB   2048      // embed
#define HB   16        // heads
#define KB   128       // head dim
#define MTOT 4096      // B*S
#define NQKV 6144      // 2*K*H + V*H
#define EPSF 1e-5f

typedef __attribute__((ext_vector_type(16))) _Float16 v16h;
typedef __attribute__((ext_vector_type(8)))  _Float16 v8h;
typedef __attribute__((ext_vector_type(8)))  float    v8f;

// D = A(16x32 f16) * B(32x16 f16) + C(16x16 f32)
__device__ __forceinline__ v8f wmma_f16(v16h a, v16h b, v8f c) {
  return __builtin_amdgcn_wmma_f32_16x16x32_f16(false, a, false, b, (short)0, c,
                                                false, false);
}

// Pack two contiguous 8-half chunks into a 16-half fragment register set.
__device__ __forceinline__ v16h pack16(const _Float16* lo, const _Float16* hi) {
  v8h l = *(const v8h*)lo;
  v8h h = *(const v8h*)hi;
  v16h r;
#pragma unroll
  for (int i = 0; i < 8; ++i) { r[i] = l[i]; r[i + 8] = h[i]; }
  return r;
}

// Async global -> LDS copy (16 bytes/lane), tracked by ASYNCcnt.
// lds_addr: per-lane LDS byte address (low 32 bits of the generic pointer).
__device__ __forceinline__ void async_copy_b128(unsigned lds_addr,
                                                const void* gsrc) {
  asm volatile("global_load_async_to_lds_b128 %0, %1, off"
               :: "v"(lds_addr), "v"((unsigned long long)(size_t)gsrc)
               : "memory");
}
__device__ __forceinline__ void wait_asynccnt0() {
  asm volatile("s_wait_asynccnt 0x0" ::: "memory");
}

// ---------------------------------------------------------------------------
// f32 -> f16 elementwise (weights)
// ---------------------------------------------------------------------------
__global__ __launch_bounds__(256) void cvt_kernel(const float* __restrict__ src,
                                                  _Float16* __restrict__ dst,
                                                  int n) {
  int i = blockIdx.x * 256 + threadIdx.x;
  if (i < n) dst[i] = (_Float16)src[i];
}

// ---------------------------------------------------------------------------
// RMSNorm: one row (B*S) per block, 256 threads, write f16
// ---------------------------------------------------------------------------
__global__ __launch_bounds__(256) void rmsnorm_kernel(const float* __restrict__ x,
                                                      const float* __restrict__ w,
                                                      _Float16* __restrict__ o) {
  const int row = blockIdx.x;
  const int tid = threadIdx.x;
  const float* xr = x + (size_t)row * EB;
  float ss = 0.f;
#pragma unroll
  for (int i = 0; i < 8; ++i) { float v = xr[tid + i * 256]; ss += v * v; }
#pragma unroll
  for (int off = 16; off > 0; off >>= 1) ss += __shfl_xor(ss, off, 32);
  __shared__ float red[8];
  if ((tid & 31) == 0) red[tid >> 5] = ss;
  __syncthreads();
  float tot = 0.f;
#pragma unroll
  for (int i = 0; i < 8; ++i) tot += red[i];
  const float rs = rsqrtf(tot * (1.0f / (float)EB) + EPSF);
  _Float16* orow = o + (size_t)row * EB;
#pragma unroll
  for (int i = 0; i < 8; ++i) {
    int e = tid + i * 256;
    orow[e] = (_Float16)(xr[e] * rs * w[e]);
  }
}

// ---------------------------------------------------------------------------
// Tiled WMMA GEMM:  C[M x N] = A[M x Kd] * B[Kd x N]   (f16 in, f32 acc)
// Block tile 128(M) x 64(N), BK=32, 8 waves: wave w -> rows [16w,16w+16).
// A tile staged via GLOBAL_LOAD_ASYNC_TO_LDS (ASYNCcnt); B tile transposed
// through VGPRs (async can't transpose).
// MODE 0: QKV epilogue (scatter q/k/v f16 + present_k/v f32)
// MODE 1: out-proj epilogue (+ residual, f32 to d_out)
// ---------------------------------------------------------------------------
template <int MODE>
__global__ __launch_bounds__(256) void gemm_kernel(
    const _Float16* __restrict__ A, const _Float16* __restrict__ Bm,
    int N, int Kd,
    _Float16* __restrict__ qh, _Float16* __restrict__ kh, _Float16* __restrict__ vh,
    float* __restrict__ pk, float* __restrict__ pv,
    const float* __restrict__ emb, float* __restrict__ outp) {
  const int tid  = threadIdx.x;
  const int wave = tid >> 5, lane = tid & 31;
  const int ln = lane & 15, lh = lane >> 4;
  const int blockM = blockIdx.y * 128, blockN = blockIdx.x * 64;

  __shared__ _Float16 ldsA[128 * 32];   // [row][k] row-major
  __shared__ _Float16 ldsBT[64 * 32];   // [n][k] (transposed)

  v8f acc[4] = {};

  // staging coordinates (whole block cooperates)
  const int arow = tid >> 1, aseg = (tid & 1) * 16;      // A: 16 halfs/thread
  const int bkd  = tid >> 3, bn0  = (tid & 7) * 8;       // B: 8 halfs/thread
  const unsigned ldsA_addr = (unsigned)(size_t)(&ldsA[arow * 32 + aseg]);

  for (int k0 = 0; k0 < Kd; k0 += 32) {
    // ---- stage A tile (128 x 32) asynchronously, no VGPR round-trip ----
    {
      const _Float16* src = A + (size_t)(blockM + arow) * Kd + k0 + aseg;
      __builtin_prefetch(src + 64, 0, 0);
      async_copy_b128(ldsA_addr,      src);
      async_copy_b128(ldsA_addr + 16, src + 8);
    }
    // ---- stage B tile transposed (32 x 64 -> [n][k]) ----
    {
      const _Float16* src = Bm + (size_t)(k0 + bkd) * N + blockN + bn0;
      __builtin_prefetch(src + (size_t)32 * N, 0, 0);
      v8h bv = *(const v8h*)src;
#pragma unroll
      for (int i = 0; i < 8; ++i) ldsBT[(bn0 + i) * 32 + bkd] = bv[i];
    }
    wait_asynccnt0();
    __syncthreads();

    // A fragment: lane ln -> row, elems: kd = lh*8 + e (e<8) / 16 + lh*8 + e
    const _Float16* ap = &ldsA[(wave * 16 + ln) * 32 + lh * 8];
    v16h afr = pack16(ap, ap + 16);
#pragma unroll
    for (int t = 0; t < 4; ++t) {
      // B fragment: lane ln -> col, elems: kd = lh*8 + e (e<8) / 16 + lh*8 + e
      const _Float16* bp = &ldsBT[(t * 16 + ln) * 32 + lh * 8];
      v16h bfr = pack16(bp, bp + 16);
      acc[t] = wmma_f16(afr, bfr, acc[t]);
    }
    __syncthreads();
  }

  // ---- epilogue ----
#pragma unroll
  for (int t = 0; t < 4; ++t) {
#pragma unroll
    for (int r = 0; r < 8; ++r) {
      const int m = blockM + wave * 16 + r + 8 * lh;   // C/D layout: M = r + 8*half
      const int c = blockN + t * 16 + ln;              // N = lane % 16
      const float val = acc[t][r];
      if (MODE == 0) {
        const int bb = m >> 11, s = m & (SB - 1);
        if (c < 2048) {                 // Q
          const int hh = c >> 7, d = c & 127;
          qh[(((size_t)(bb * HB + hh)) * SB + s) * KB + d] = (_Float16)val;
        } else if (c < 4096) {          // K (also present_k, pre-rotation)
          const int c2 = c - 2048, hh = c2 >> 7, d = c2 & 127;
          const size_t idx = (((size_t)(bb * HB + hh)) * SB + s) * KB + d;
          kh[idx] = (_Float16)val;
          pk[idx] = val;
        } else {                        // V (also present_v)
          const int c2 = c - 4096, hh = c2 >> 7, d = c2 & 127;
          const size_t idx = (((size_t)(bb * HB + hh)) * SB + s) * KB + d;
          vh[idx] = (_Float16)val;
          pv[idx] = val;
        }
      } else {
        const size_t idx = (size_t)m * EB + c;
        outp[idx] = val + emb[idx];
      }
    }
  }
}

// ---------------------------------------------------------------------------
// RoPE in place on q_f16 and k_f16 ([B,H,S,K]); each thread owns (d, d+64).
// cos/sin buffers: (2,1,1,S,K); [0] for q, [1] for k.
// ---------------------------------------------------------------------------
__global__ __launch_bounds__(256) void rope_kernel(_Float16* __restrict__ q,
                                                   _Float16* __restrict__ k,
                                                   const float* __restrict__ cosb,
                                                   const float* __restrict__ sinb) {
  const size_t idx = (size_t)blockIdx.x * 256 + threadIdx.x;  // B*H*S*64 threads
  const int d = (int)(idx & 63);
  const size_t row = idx >> 6;                 // (b*H + h)*S + s
  const int s = (int)(row & (SB - 1));
  const size_t base = row * KB;
  const size_t a0 = (size_t)s * KB + d, a1 = a0 + 64;

  { // queries: buffer [0]
    const float c0 = cosb[a0], c1 = cosb[a1];
    const float s0 = sinb[a0], s1 = sinb[a1];
    const float x0 = (float)q[base + d], x1 = (float)q[base + d + 64];
    q[base + d]      = (_Float16)(c0 * x0 + s0 * x1);
    q[base + d + 64] = (_Float16)(c1 * x1 + s1 * x0);
  }
  { // keys: buffer [1] at offset S*K
    const size_t off = (size_t)SB * KB;
    const float c0 = cosb[off + a0], c1 = cosb[off + a1];
    const float s0 = sinb[off + a0], s1 = sinb[off + a1];
    const float x0 = (float)k[base + d], x1 = (float)k[base + d + 64];
    k[base + d]      = (_Float16)(c0 * x0 + s0 * x1);
    k[base + d + 64] = (_Float16)(c1 * x1 + s1 * x0);
  }
}

// ---------------------------------------------------------------------------
// Causal flash attention. Grid (S/64, H, B); 4 waves/block, 16 queries/wave.
// K tile staged via async global->LDS; V transposed through VGPRs.
// Online softmax in f32; P transposed to A-fragment layout via per-wave LDS.
// All control flow is wave-uniform -> EXEC stays all-1s around WMMA.
// ---------------------------------------------------------------------------
__global__ __launch_bounds__(128) void attn_kernel(const _Float16* __restrict__ qf,
                                                   const _Float16* __restrict__ kf,
                                                   const _Float16* __restrict__ vf,
                                                   _Float16* __restrict__ attn) {
  const int b = blockIdx.z, h = blockIdx.y;
  const int tid = threadIdx.x, wave = tid >> 5, lane = tid & 31;
  const int ln = lane & 15, lh = lane >> 4;
  const int qbase = blockIdx.x * 64 + wave * 16;
  const size_t bh = (size_t)(b * HB + h);

  __shared__ _Float16 ldsK[32 * 128];       // [key][kd]
  __shared__ _Float16 ldsVT[128 * 32];      // [vdim][key]
  __shared__ _Float16 ldsP[4][16 * 32];     // per-wave P scratch [m][key]

  // ---- load Q fragments (16 x 128 in 4 chunks of 32) ----
  v16h qfrag[4];
  {
    const _Float16* qrow = qf + (bh * SB + qbase + ln) * KB;
#pragma unroll
    for (int c = 0; c < 4; ++c) {
      const _Float16* p = qrow + c * 32 + lh * 8;
      qfrag[c] = pack16(p, p + 16);
    }
  }

  const v8f vzero = {};
  float mrow[8], lrow[8];
  v8f oacc[8];
#pragma unroll
  for (int r = 0; r < 8; ++r) { mrow[r] = -1e30f; lrow[r] = 0.f; }
#pragma unroll
  for (int vc = 0; vc < 8; ++vc) oacc[vc] = vzero;

  const float scale = 0.08838834764831845f;   // 1/sqrt(128)
  const int npairs = (blockIdx.x + 1) * 2;    // causal: keys up to block diagonal

  // per-thread staging coordinates (32 keys x 128 dims, 128 threads)
  const int skey = tid >> 2, sseg = tid & 3;  // 32 halfs per thread each
  const unsigned ldsK_addr =
      (unsigned)(size_t)(&ldsK[skey * 128 + sseg * 32]);

  for (int jp = 0; jp < npairs; ++jp) {
    const int kbase = jp * 32;
    // ---- K tile: async global -> LDS (row-major, 64B/thread) ----
    {
      const _Float16* ksrc = kf + (bh * SB + kbase + skey) * KB + sseg * 32;
      async_copy_b128(ldsK_addr,      ksrc);
      async_copy_b128(ldsK_addr + 16, ksrc + 8);
      async_copy_b128(ldsK_addr + 32, ksrc + 16);
      async_copy_b128(ldsK_addr + 48, ksrc + 24);
    }
    // ---- V tile: transpose through VGPRs into [vdim][key] ----
    {
      const _Float16* vsrc = vf + (bh * SB + kbase + skey) * KB + sseg * 32;
#pragma unroll
      for (int i = 0; i < 32; ++i)
        ldsVT[(sseg * 32 + i) * 32 + skey] = vsrc[i];
    }
    wait_asynccnt0();
    __syncthreads();

    if (kbase <= qbase + 15) {                 // wave-uniform branch
      // ---- scores: Q(16x128) . K^T -> two 16x16 tiles ----
      v8f s0 = {}, s1 = {};
#pragma unroll
      for (int c = 0; c < 4; ++c) {
        const _Float16* p0 = &ldsK[ln * 128 + c * 32 + lh * 8];
        const _Float16* p1 = &ldsK[(16 + ln) * 128 + c * 32 + lh * 8];
        v16h b0 = pack16(p0, p0 + 16);
        v16h b1 = pack16(p1, p1 + 16);
        s0 = wmma_f16(qfrag[c], b0, s0);
        s1 = wmma_f16(qfrag[c], b1, s1);
      }
      // ---- mask + online softmax (row reductions within 16-lane halves) ----
      float corr[8];
#pragma unroll
      for (int r = 0; r < 8; ++r) {
        const int qi = qbase + r + 8 * lh;
        float a0 = (kbase + ln      <= qi) ? s0[r] * scale : -1e30f;
        float a1 = (kbase + 16 + ln <= qi) ? s1[r] * scale : -1e30f;
        float rm = fmaxf(a0, a1);
#pragma unroll
        for (int off = 1; off < 16; off <<= 1)
          rm = fmaxf(rm, __shfl_xor(rm, off, 32));
        const float mnew = fmaxf(mrow[r], rm);
        const float p0 = __expf(a0 - mnew);
        const float p1 = __expf(a1 - mnew);
        float rs = p0 + p1;
#pragma unroll
        for (int off = 1; off < 16; off <<= 1)
          rs += __shfl_xor(rs, off, 32);
        corr[r] = __expf(mrow[r] - mnew);
        lrow[r] = lrow[r] * corr[r] + rs;
        mrow[r] = mnew;
        const int m = r + 8 * lh;
        ldsP[wave][m * 32 + ln]      = (_Float16)p0;
        ldsP[wave][m * 32 + 16 + ln] = (_Float16)p1;
      }
      // ---- transpose P through LDS into A-fragment layout ----
      const _Float16* pp = &ldsP[wave][ln * 32 + lh * 8];
      v16h pfr = pack16(pp, pp + 16);
      // ---- O = diag(corr) * O + P * V  (8 N-chunks of 16) ----
#pragma unroll
      for (int vc = 0; vc < 8; ++vc) {
#pragma unroll
        for (int r = 0; r < 8; ++r) oacc[vc][r] *= corr[r];
        const _Float16* vp = &ldsVT[(vc * 16 + ln) * 32 + lh * 8];
        v16h vfr = pack16(vp, vp + 16);
        oacc[vc] = wmma_f16(pfr, vfr, oacc[vc]);
      }
    }
    __syncthreads();
  }

  // ---- normalize and write (B, S, H*V) f16 for the output GEMM ----
  float invl[8];
#pragma unroll
  for (int r = 0; r < 8; ++r) invl[r] = 1.0f / lrow[r];
#pragma unroll
  for (int vc = 0; vc < 8; ++vc) {
#pragma unroll
    for (int r = 0; r < 8; ++r) {
      const int s = qbase + r + 8 * lh;
      attn[((size_t)b * SB + s) * (HB * KB) + h * KB + vc * 16 + ln] =
          (_Float16)(oacc[vc][r] * invl[r]);
    }
  }
}

// ---------------------------------------------------------------------------
// Host-side launcher
// ---------------------------------------------------------------------------
extern "C" void kernel_launch(void* const* d_in, const int* in_sizes, int n_in,
                              void* d_out, int out_size, void* d_ws, size_t ws_size,
                              hipStream_t stream) {
  (void)in_sizes; (void)n_in; (void)out_size; (void)ws_size;
  const float* emb   = (const float*)d_in[0];
  const float* cosb  = (const float*)d_in[1];
  const float* sinb  = (const float*)d_in[2];
  /* d_in[3] = causal_buffer: causality computed analytically */
  const float* wnorm = (const float*)d_in[4];
  const float* wqkv  = (const float*)d_in[5];
  const float* wout  = (const float*)d_in[6];

  float* outp = (float*)d_out;                 // (B,S,E)
  float* pk   = outp + (size_t)8388608;        // present_k (B,H,S,K)
  float* pv   = outp + (size_t)16777216;       // present_v (B,H,S,V)

  _Float16* ws      = (_Float16*)d_ws;
  _Float16* normed  = ws;                       //  4096*2048
  _Float16* wqkv_h  = ws + (size_t)8388608;     //  2048*6144
  _Float16* wout_h  = ws + (size_t)20971520;    //  2048*2048
  _Float16* qh      = ws + (size_t)25165824;    //  B,H,S,K
  _Float16* kh      = ws + (size_t)33554432;
  _Float16* vh      = ws + (size_t)41943040;
  _Float16* attnh   = ws + (size_t)50331648;    //  B,S,H*V

  // 1. weights -> f16
  cvt_kernel<<<(12582912 + 255) / 256, 256, 0, stream>>>(wqkv, wqkv_h, 12582912);
  cvt_kernel<<<(4194304 + 255) / 256, 256, 0, stream>>>(wout, wout_h, 4194304);
  // 2. RMSNorm -> f16
  rmsnorm_kernel<<<MTOT, 256, 0, stream>>>(emb, wnorm, normed);
  // 3. QKV GEMM (M=4096, N=6144, K=2048) with scatter epilogue
  gemm_kernel<0><<<dim3(NQKV / 64, MTOT / 128), 256, 0, stream>>>(
      normed, wqkv_h, NQKV, EB, qh, kh, vh, pk, pv, nullptr, nullptr);
  // 4. RoPE in place on q, k
  rope_kernel<<<(2 * HB * SB * 64) / 256, 256, 0, stream>>>(qh, kh, cosb, sinb);
  // 5. causal flash attention
  attn_kernel<<<dim3(SB / 64, HB, 2), 128, 0, stream>>>(qh, kh, vh, attnh);
  // 6. output GEMM (M=4096, N=2048, K=2048) + residual
  gemm_kernel<1><<<dim3(EB / 64, MTOT / 128), 256, 0, stream>>>(
      attnh, wout_h, EB, EB, nullptr, nullptr, nullptr, nullptr, nullptr,
      emb, outp);
}
